// InputProjection_50689204027588
// MI455X (gfx1250) — matrix operand order repaired and checked by reading, not compiled, not run on hardware
//
#include <hip/hip_runtime.h>

typedef __attribute__((ext_vector_type(16))) _Float16 v16h;
typedef __attribute__((ext_vector_type(8)))  _Float16 v8h;
typedef __attribute__((ext_vector_type(4)))  _Float16 v4h;
typedef __attribute__((ext_vector_type(8)))  float    v8f;

#define HID   1024
#define KVD   256
#define NTOT  1536           // 1024 (Q) + 256 (K) + 256 (V)
#define SEQ   4096
#define BATCH 8
#define MTOT  (BATCH * SEQ)  // 32768 tokens
#define EPS   1e-5f

// ---------------------------------------------------------------------------
// Kernel 1: fused LayerNorm, fp32 in -> f16 out (one block per token row).
// 256 threads x 4 floats = 1024 columns; b128 loads, wave32 shuffle reduce.
// ---------------------------------------------------------------------------
__global__ void __launch_bounds__(256)
ln_f16_kernel(const float* __restrict__ x,
              const float* __restrict__ gamma,
              const float* __restrict__ beta,
              _Float16* __restrict__ xn)
{
    const int row = blockIdx.x;
    const int tid = threadIdx.x;
    const float* xr = x + (size_t)row * HID;

    float4 v = ((const float4*)xr)[tid];
    float s  = v.x + v.y + v.z + v.w;
    float sq = v.x * v.x + v.y * v.y + v.z * v.z + v.w * v.w;

    // wave32 reduction
    #pragma unroll
    for (int off = 16; off > 0; off >>= 1) {
        s  += __shfl_down(s,  off, 32);
        sq += __shfl_down(sq, off, 32);
    }

    __shared__ float ssum[8], ssq[8];
    __shared__ float smu, srstd;
    if ((tid & 31) == 0) { ssum[tid >> 5] = s; ssq[tid >> 5] = sq; }
    __syncthreads();
    if (tid == 0) {
        float S = 0.f, Q = 0.f;
        #pragma unroll
        for (int i = 0; i < 8; ++i) { S += ssum[i]; Q += ssq[i]; }
        float mu  = S * (1.0f / HID);
        float var = Q * (1.0f / HID) - mu * mu;
        smu   = mu;
        srstd = rsqrtf(var + EPS);
    }
    __syncthreads();
    const float mu = smu, r = srstd;

    float4 g = ((const float4*)gamma)[tid];
    float4 b = ((const float4*)beta)[tid];
    v4h h;
    h[0] = (_Float16)((v.x - mu) * r * g.x + b.x);
    h[1] = (_Float16)((v.y - mu) * r * g.y + b.y);
    h[2] = (_Float16)((v.z - mu) * r * g.z + b.z);
    h[3] = (_Float16)((v.w - mu) * r * g.w + b.w);
    ((v4h*)(xn + (size_t)row * HID))[tid] = h;
}

// ---------------------------------------------------------------------------
// Kernel 2: pack Q/K/V weights into one concatenated f16 matrix [1536, 1024].
// ---------------------------------------------------------------------------
__global__ void __launch_bounds__(256)
wconv_kernel(const float* __restrict__ qw,
             const float* __restrict__ kw,
             const float* __restrict__ vw,
             _Float16* __restrict__ wf)
{
    const size_t i = (size_t)blockIdx.x * 256 + threadIdx.x; // < 1536*1024
    const int n = (int)(i >> 10);
    float val;
    if (n < HID)            val = qw[i];
    else if (n < HID + KVD) val = kw[i - (size_t)HID * HID];
    else                    val = vw[i - (size_t)(HID + KVD) * HID];
    wf[i] = (_Float16)val;
}

// ---------------------------------------------------------------------------
// Epilogue: write one 16x16 C tile to the GQA-permuted q/k/v output + bias.
// C layout (32-bit, 16x16): lane<16 -> N=lane, VGPR r -> M=r;
//                           lane>=16 -> N=lane-16, VGPR r -> M=r+8.
// ---------------------------------------------------------------------------
__device__ __forceinline__ void store_tile(float* __restrict__ out,
                                           const float* __restrict__ qb,
                                           const float* __restrict__ kb,
                                           const float* __restrict__ vb,
                                           v8f acc, int m0, int lane, int ntile)
{
    const size_t QSZ  = (size_t)BATCH * 16 * SEQ * 64; // 33,554,432
    const size_t KVSZ = (size_t)BATCH * 4  * SEQ * 64; //  8,388,608
    const int mbase = m0 + ((lane >> 4) << 3);
    const int n = ntile + (lane & 15);

    if (n < HID) {                         // Q: [b, h, s, d]
        const int h = n >> 6, d = n & 63;
        const float bias = qb[n];
        #pragma unroll
        for (int r = 0; r < 8; ++r) {
            const int m = mbase + r;
            const int b = m >> 12, s = m & (SEQ - 1);
            const size_t off = (((size_t)(b * 16 + h)) << 18) |
                               ((size_t)s << 6) | (size_t)d;
            out[off] = acc[r] + bias;
        }
    } else if (n < HID + KVD) {            // K: [b, 1, g*SEQ + s, d]
        const int g = (n - HID) >> 6, d = (n - HID) & 63;
        const float bias = kb[n - HID];
        #pragma unroll
        for (int r = 0; r < 8; ++r) {
            const int m = mbase + r;
            const int b = m >> 12, s = m & (SEQ - 1);
            const size_t off = QSZ + ((size_t)b << 20) +
                               (((size_t)(g * SEQ + s)) << 6) + (size_t)d;
            out[off] = acc[r] + bias;
        }
    } else {                               // V: [b, 1, g*SEQ + s, d]
        const int g = (n - HID - KVD) >> 6, d = (n - HID - KVD) & 63;
        const float bias = vb[n - HID - KVD];
        #pragma unroll
        for (int r = 0; r < 8; ++r) {
            const int m = mbase + r;
            const int b = m >> 12, s = m & (SEQ - 1);
            const size_t off = QSZ + KVSZ + ((size_t)b << 20) +
                               (((size_t)(g * SEQ + s)) << 6) + (size_t)d;
            out[off] = acc[r] + bias;
        }
    }
}

// ---------------------------------------------------------------------------
// Kernel 3: WMMA GEMM  C[32768,1536] = xn_f16 @ W_f16^T  (NT, both K-major).
// Block = 256 threads = 8 waves arranged 2 (M) x 4 (N).
// Wave tile: M=16, N=64 (4x v_wmma_f32_16x16x32_f16 per K-step of 32).
// Block tile: 32 x 256. Grid: (1024, 6).
// ---------------------------------------------------------------------------
__global__ void __launch_bounds__(256)
qkv_gemm_kernel(const _Float16* __restrict__ xn,
                const _Float16* __restrict__ wf,
                const float* __restrict__ qb,
                const float* __restrict__ kb,
                const float* __restrict__ vb,
                float* __restrict__ out)
{
    const int lane = threadIdx.x & 31;
    const int wid  = threadIdx.x >> 5;
    const int wm   = wid >> 2;      // 0..1
    const int wn   = wid & 3;       // 0..3
    const int m0   = blockIdx.x * 32 + wm * 16;
    const int n0   = blockIdx.y * 256 + wn * 64;

    // A fragment addressing: lane<16 -> row m0+lane, K 0..15;
    //                        lane>=16 -> row m0+lane-16, K 16..31.
    const _Float16* __restrict__ abase =
        xn + (size_t)(m0 + (lane & 15)) * HID + ((lane >> 4) << 4);
    // B fragment addressing: lane<16 -> col n+lane, K 0..15; lane>=16 -> K 16..31.
    const size_t bk = (size_t)((lane >> 4) << 4);
    const _Float16* __restrict__ b0p = wf + (size_t)(n0 +  0 + (lane & 15)) * HID + bk;
    const _Float16* __restrict__ b1p = wf + (size_t)(n0 + 16 + (lane & 15)) * HID + bk;
    const _Float16* __restrict__ b2p = wf + (size_t)(n0 + 32 + (lane & 15)) * HID + bk;
    const _Float16* __restrict__ b3p = wf + (size_t)(n0 + 48 + (lane & 15)) * HID + bk;

    v8f acc0 = {}, acc1 = {}, acc2 = {}, acc3 = {};

    #pragma unroll 4
    for (int k0 = 0; k0 < HID; k0 += 32) {
        v8h alo = *(const v8h*)(abase + k0);
        v8h ahi = *(const v8h*)(abase + k0 + 8);
        v16h a = __builtin_shufflevector(alo, ahi,
                 0,1,2,3,4,5,6,7,8,9,10,11,12,13,14,15);

        v8h l0 = *(const v8h*)(b0p + k0), h0 = *(const v8h*)(b0p + k0 + 8);
        v8h l1 = *(const v8h*)(b1p + k0), h1 = *(const v8h*)(b1p + k0 + 8);
        v8h l2 = *(const v8h*)(b2p + k0), h2 = *(const v8h*)(b2p + k0 + 8);
        v8h l3 = *(const v8h*)(b3p + k0), h3 = *(const v8h*)(b3p + k0 + 8);
        v16h b0v = __builtin_shufflevector(l0, h0, 0,1,2,3,4,5,6,7,8,9,10,11,12,13,14,15);
        v16h b1v = __builtin_shufflevector(l1, h1, 0,1,2,3,4,5,6,7,8,9,10,11,12,13,14,15);
        v16h b2v = __builtin_shufflevector(l2, h2, 0,1,2,3,4,5,6,7,8,9,10,11,12,13,14,15);
        v16h b3v = __builtin_shufflevector(l3, h3, 0,1,2,3,4,5,6,7,8,9,10,11,12,13,14,15);

        acc0 = __builtin_amdgcn_wmma_f32_16x16x32_f16(false, a, false, b0v,
                                                      (short)0, acc0, false, false);
        acc1 = __builtin_amdgcn_wmma_f32_16x16x32_f16(false, a, false, b1v,
                                                      (short)0, acc1, false, false);
        acc2 = __builtin_amdgcn_wmma_f32_16x16x32_f16(false, a, false, b2v,
                                                      (short)0, acc2, false, false);
        acc3 = __builtin_amdgcn_wmma_f32_16x16x32_f16(false, a, false, b3v,
                                                      (short)0, acc3, false, false);
    }

    store_tile(out, qb, kb, vb, acc0, m0, lane, n0 +  0);
    store_tile(out, qb, kb, vb, acc1, m0, lane, n0 + 16);
    store_tile(out, qb, kb, vb, acc2, m0, lane, n0 + 32);
    store_tile(out, qb, kb, vb, acc3, m0, lane, n0 + 48);
}

// ---------------------------------------------------------------------------
// Host-side launcher (graph-capture safe: only kernel launches on `stream`).
// Workspace: xn_f16 (32768*1024*2B = 64MB) + w_f16 (1536*1024*2B = 3MB).
// ---------------------------------------------------------------------------
extern "C" void kernel_launch(void* const* d_in, const int* in_sizes, int n_in,
                              void* d_out, int out_size, void* d_ws, size_t ws_size,
                              hipStream_t stream)
{
    const float* x   = (const float*)d_in[0];
    const float* qw  = (const float*)d_in[1];
    const float* kw  = (const float*)d_in[2];
    const float* vw  = (const float*)d_in[3];
    const float* qbs = (const float*)d_in[4];
    const float* kbs = (const float*)d_in[5];
    const float* vbs = (const float*)d_in[6];
    const float* g   = (const float*)d_in[7];
    const float* bt  = (const float*)d_in[8];

    _Float16* xn = (_Float16*)d_ws;
    _Float16* wf = xn + (size_t)MTOT * HID;

    ln_f16_kernel<<<MTOT, 256, 0, stream>>>(x, g, bt, xn);
    wconv_kernel<<<(NTOT * HID) / 256, 256, 0, stream>>>(qw, kw, vw, wf);

    dim3 grid(MTOT / 32, NTOT / 256);
    qkv_gemm_kernel<<<grid, 256, 0, stream>>>(xn, wf, qbs, kbs, vbs, (float*)d_out);
}
